// SelfAttention_2697239462114
// MI455X (gfx1250) — matrix operand order repaired and checked by reading, not compiled, not run on hardware
//
#include <hip/hip_runtime.h>
#include <hip/hip_bf16.h>

// ---------------------------------------------------------------------------
// Self-attention forward for MI455X (gfx1250), bf16 WMMA everywhere,
// async-to-LDS staging (ASYNCcnt) for the attention K/V tiles.
// B=2, L=2048, D=1024, H=16, hd=64.
// ---------------------------------------------------------------------------

#define B_SZ   2
#define L_SEQ  2048
#define D_MOD  1024
#define N_HEAD 16
#define HD     64
#define EPS_F  1.19209290e-07f

typedef __attribute__((ext_vector_type(16))) __bf16 v16bf;
typedef __attribute__((ext_vector_type(8)))  float  v8f;

// ---------------------------------------------------------------------------
// WMMA helper: D = A(16x32 bf16) * B(32x16 bf16) + C(16x16 f32)
// ---------------------------------------------------------------------------
__device__ __forceinline__ v8f wmma_bf16(v16bf a, v16bf b, v8f c) {
    return __builtin_amdgcn_wmma_f32_16x16x32_bf16(
        /*neg_a=*/false, a, /*neg_b=*/false, b,
        /*c_mod=*/(short)0, c, /*reuse_a=*/false, /*reuse_b=*/false);
}

__device__ __forceinline__ v8f vzero8() {
    v8f z;
#pragma unroll
    for (int i = 0; i < 8; ++i) z[i] = 0.0f;
    return z;
}

// ---------------------------------------------------------------------------
// A-fragment (16x32, row-major source, leading dim ld):
//   lane = m + 16*bit3(k);  e<8 -> k = 8*hi + e ; e>=8 -> k = 16 + 8*hi + (e-8)
// => per lane: 16B chunk at (m*ld + 8*hi) and 16B chunk 16 elements later.
// Works for global or LDS pointers (addrspace inferred after inlining).
// ---------------------------------------------------------------------------
__device__ __forceinline__ v16bf load_frag_a(const __bf16* __restrict__ base, int ld) {
    const int lane = threadIdx.x & 31;
    const __bf16* p = base + (size_t)(lane & 15) * ld + ((lane >> 4) << 3);
    union { v16bf v; uint4 q[2]; } u;
    u.q[0] = *reinterpret_cast<const uint4*>(p);
    u.q[1] = *reinterpret_cast<const uint4*>(p + 16);
    return u.v;
}

// ---------------------------------------------------------------------------
// B-fragment (32x16) where B = W^T, W is (N,K) row-major with leading dim ld:
//   B(k,n) = W[n][k];  lane = n + 16*bit3(k);  e<8 -> k = 8*hi + e ; e>=8 -> k = 16 + 8*hi + (e-8)
// => per lane: 16B chunk at (n*ld + 8*hi) and 16B chunk 16 elements later.
// ---------------------------------------------------------------------------
__device__ __forceinline__ v16bf load_frag_b(const __bf16* __restrict__ base, int ld) {
    const int lane = threadIdx.x & 31;
    const __bf16* p = base + (size_t)(lane & 15) * ld + ((lane >> 4) << 3);
    union { v16bf v; uint4 q[2]; } u;
    u.q[0] = *reinterpret_cast<const uint4*>(p);
    u.q[1] = *reinterpret_cast<const uint4*>(p + 16);
    return u.v;
}

// ---------------------------------------------------------------------------
// Async global->LDS 16B copy (gfx1250, ASYNCcnt-tracked, no VGPR staging).
// lds_off: wave-relative LDS byte offset (= low 32 bits of generic pointer).
// ---------------------------------------------------------------------------
__device__ __forceinline__ void async_copy_b128(unsigned lds_off, const void* gaddr) {
    asm volatile("global_load_async_to_lds_b128 %0, %1, off"
                 :: "v"(lds_off), "v"(gaddr) : "memory");
}
__device__ __forceinline__ unsigned lds_offset(const void* p) {
    return (unsigned)(uintptr_t)p;   // generic LDS addr: low 32 bits = LDS offset
}

// ---------------------------------------------------------------------------
// Kernel 1: f32 -> bf16 conversion
// ---------------------------------------------------------------------------
__global__ __launch_bounds__(256) void cvt_f32_bf16(const float* __restrict__ in,
                                                    __bf16* __restrict__ out, int n) {
    int i = blockIdx.x * 256 + threadIdx.x;
    if (i < n) out[i] = (__bf16)in[i];
}

// ---------------------------------------------------------------------------
// Kernel 2/5: C[M,N] f32 = A_bf16[M,K] * W_bf16[N,K]^T
// Block: 256 thr = 8 waves in 2(M) x 4(N); block tile 32 x 256;
// wave tile 16 x 64.  All 5 fragments loaded before the 4 WMMAs so the
// scheduler can overlap loads with matrix ops (partial loadcnt waits).
// Grid: (N/256, M/32).
// ---------------------------------------------------------------------------
__global__ __launch_bounds__(256) void gemm_bf16_nt(const __bf16* __restrict__ A,
                                                    const __bf16* __restrict__ W,
                                                    float* __restrict__ C,
                                                    int M, int N, int K) {
    const int wid  = threadIdx.x >> 5;
    const int lane = threadIdx.x & 31;
    const int m0 = blockIdx.y * 32 + (wid >> 2) * 16;
    const int n0 = blockIdx.x * 256 + (wid & 3) * 64;

    v8f acc[4];
#pragma unroll
    for (int t = 0; t < 4; ++t) acc[t] = vzero8();

    const __bf16* Arow = A + (size_t)m0 * K;
    for (int k0 = 0; k0 < K; k0 += 32) {
        if (k0 + 64 < K) {
            __builtin_prefetch(Arow + k0 + 64, 0, 1);      // global_prefetch_b8
        }
        v16bf a = load_frag_a(Arow + k0, K);
        v16bf b[4];
#pragma unroll
        for (int t = 0; t < 4; ++t)
            b[t] = load_frag_b(W + (size_t)(n0 + 16 * t) * K + k0, K);
#pragma unroll
        for (int t = 0; t < 4; ++t)
            acc[t] = wmma_bf16(a, b[t], acc[t]);
    }

    const int row  = m0 + ((lane >> 4) << 3);
    const int col0 = lane & 15;
#pragma unroll
    for (int t = 0; t < 4; ++t)
#pragma unroll
        for (int v = 0; v < 8; ++v)
            C[(size_t)(row + v) * N + n0 + 16 * t + col0] = acc[t][v];
}

// ---------------------------------------------------------------------------
// Kernel 3: RMSNorm(q,k) + scatter to attention layouts (bf16).
//  qkv f32 laid out (B*L, 3, H, hd).  Outputs:
//   Q[bh][l][d] (1/sqrt(hd) folded in), K[bh][l][d], Vt[bh][d][l].
//  grid.x = B*L; 256 threads: thread t -> head h = t>>4, d0 = (t&15)*4.
// ---------------------------------------------------------------------------
__global__ __launch_bounds__(256) void rmsnorm_scatter(const float* __restrict__ qkv,
                                                       const float* __restrict__ w_qn,
                                                       const float* __restrict__ w_kn,
                                                       __bf16* __restrict__ Qo,
                                                       __bf16* __restrict__ Ko,
                                                       __bf16* __restrict__ Vt) {
    const int bl = blockIdx.x;               // b*L + l
    const int b  = bl >> 11;
    const int l  = bl & (L_SEQ - 1);
    const int t  = threadIdx.x;
    const int h  = t >> 4;
    const int d0 = (t & 15) * 4;

    const float* base = qkv + (size_t)bl * (3 * D_MOD) + h * HD + d0;
    float4 q = *reinterpret_cast<const float4*>(base);
    float4 k = *reinterpret_cast<const float4*>(base + D_MOD);
    float4 v = *reinterpret_cast<const float4*>(base + 2 * D_MOD);

    float sq = q.x * q.x + q.y * q.y + q.z * q.z + q.w * q.w;
    float sk = k.x * k.x + k.y * k.y + k.z * k.z + k.w * k.w;
#pragma unroll
    for (int m = 1; m < 16; m <<= 1) {       // reduce within 16-lane head group
        sq += __shfl_xor(sq, m, 32);
        sk += __shfl_xor(sk, m, 32);
    }
    const float rq = rsqrtf(sq * (1.0f / HD) + EPS_F) * 0.125f; // fold 1/sqrt(64)
    const float rk = rsqrtf(sk * (1.0f / HD) + EPS_F);

    const size_t bh = (size_t)b * N_HEAD + h;
    __bf16* qo = Qo + (bh * L_SEQ + l) * HD + d0;
    __bf16* ko = Ko + (bh * L_SEQ + l) * HD + d0;
    qo[0] = (__bf16)(q.x * rq * w_qn[d0 + 0]);
    qo[1] = (__bf16)(q.y * rq * w_qn[d0 + 1]);
    qo[2] = (__bf16)(q.z * rq * w_qn[d0 + 2]);
    qo[3] = (__bf16)(q.w * rq * w_qn[d0 + 3]);
    ko[0] = (__bf16)(k.x * rk * w_kn[d0 + 0]);
    ko[1] = (__bf16)(k.y * rk * w_kn[d0 + 1]);
    ko[2] = (__bf16)(k.z * rk * w_kn[d0 + 2]);
    ko[3] = (__bf16)(k.w * rk * w_kn[d0 + 3]);

    __bf16* vt = Vt + (bh * HD + d0) * L_SEQ + l;   // transposed store
    vt[0 * L_SEQ] = (__bf16)v.x;
    vt[1 * L_SEQ] = (__bf16)v.y;
    vt[2 * L_SEQ] = (__bf16)v.z;
    vt[3 * L_SEQ] = (__bf16)v.w;
}

// ---------------------------------------------------------------------------
// Kernel 4: flash attention for one (b,h).  Block = 8 waves, each wave owns
// 16 q rows (block = 128 rows).  The 32-key K tile (4KB, contiguous) and the
// V tile (4KB) are shared by all 8 waves, so they are staged into LDS with
// async global->LDS copies, double-buffered:
//   issue(i+1) -> s_wait_asynccnt(fill i done) -> barrier -> compute(i) -> barrier
// QK^T: 4 WMMAs; online softmax; P staged via wave-private LDS; PV: 4 WMMAs.
// Grid: (L/128, B*H).
// ---------------------------------------------------------------------------
__global__ __launch_bounds__(256) void flash_attn(const __bf16* __restrict__ Q,
                                                  const __bf16* __restrict__ Kt,
                                                  const __bf16* __restrict__ Vt,
                                                  __bf16* __restrict__ O) {
    __shared__ __align__(16) __bf16 Ksh[2][32 * HD];   // [key][d]   8KB
    __shared__ __align__(16) __bf16 Vsh[2][HD * 32];   // [d][key]   8KB
    __shared__ __align__(16) __bf16 pbuf[8][16 * 32];  // wave-private P tiles

    const int tid  = threadIdx.x;
    const int wid  = tid >> 5;
    const int lane = tid & 31;
    const int bh   = blockIdx.y;
    const int m0   = blockIdx.x * 128 + wid * 16;

    const __bf16* Qb = Q  + ((size_t)bh * L_SEQ + m0) * HD;
    const __bf16* Kb = Kt + (size_t)bh * L_SEQ * HD;
    const __bf16* Vb = Vt + (size_t)bh * HD * L_SEQ;

    // Q fragments held for the whole pass (K-dim 64 = 2 frags)
    const v16bf qa0 = load_frag_a(Qb, HD);
    const v16bf qa1 = load_frag_a(Qb + 32, HD);

    v8f acc[4];
#pragma unroll
    for (int t = 0; t < 4; ++t) acc[t] = vzero8();
    float m_r[8], l_r[8];
#pragma unroll
    for (int v = 0; v < 8; ++v) { m_r[v] = -1e30f; l_r[v] = 0.0f; }

    __bf16* myp = &pbuf[wid][0];
    const int hi  = lane >> 4;
    const int col = lane & 15;

    // async stage of one 32-key K/V tile: 256 threads x 16B each per tensor
    const int vrow = tid >> 2;          // 0..63 (d)
    const int vseg = tid & 3;           // 4 x 16B per V row
    auto stage_kv = [&](int j0, int buf) {
        // K tile: rows j0..j0+31, 64 bf16 each -> 4096B fully contiguous
        async_copy_b128(lds_offset(&Ksh[buf][tid * 8]),
                        (const void*)(Kb + (size_t)j0 * HD + tid * 8));
        // V tile: 64 rows of 32 keys (64B) from Vt, stride L
        async_copy_b128(lds_offset(&Vsh[buf][vrow * 32 + vseg * 8]),
                        (const void*)(Vb + (size_t)vrow * L_SEQ + j0 + vseg * 8));
    };

    stage_kv(0, 0);                                     // prologue fill

    for (int j0 = 0; j0 < L_SEQ; j0 += 32) {
        const int cur = (j0 >> 5) & 1;
        if (j0 + 32 < L_SEQ) {
            stage_kv(j0 + 32, cur ^ 1);                 // fill next buffer
            asm volatile("s_wait_asynccnt 0x2" ::: "memory");  // fill(cur) done
        } else {
            asm volatile("s_wait_asynccnt 0x0" ::: "memory");
        }
        __syncthreads();                                // all fills visible

        const __bf16* kc = &Ksh[cur][0];
        const __bf16* vc = &Vsh[cur][0];

        // ---- scores: 16 x 32 tile = two 16x16 WMMA outputs -------------
        v16bf kb[4];
        kb[0] = load_frag_b(kc, 64);                    // keys 0-15, d 0-31
        kb[1] = load_frag_b(kc + 32, 64);               // keys 0-15, d 32-63
        kb[2] = load_frag_b(kc + 16 * 64, 64);          // keys 16-31, d 0-31
        kb[3] = load_frag_b(kc + 16 * 64 + 32, 64);     // keys 16-31, d 32-63
        v8f s0 = vzero8(), s1 = vzero8();
        s0 = wmma_bf16(qa0, kb[0], s0);
        s0 = wmma_bf16(qa1, kb[1], s0);
        s1 = wmma_bf16(qa0, kb[2], s1);
        s1 = wmma_bf16(qa1, kb[3], s1);

        // ---- online softmax (row stats replicated across 16-lane halves)
#pragma unroll
        for (int v = 0; v < 8; ++v) {
            float mx = fmaxf(s0[v], s1[v]);
#pragma unroll
            for (int m = 1; m < 16; m <<= 1) mx = fmaxf(mx, __shfl_xor(mx, m, 32));
            const float mn = fmaxf(m_r[v], mx);
            const float sc = __expf(m_r[v] - mn);
            m_r[v] = mn;
            const float p0 = __expf(s0[v] - mn);
            const float p1 = __expf(s1[v] - mn);
            float rs = p0 + p1;
#pragma unroll
            for (int m = 1; m < 16; m <<= 1) rs += __shfl_xor(rs, m, 32);
            l_r[v] = l_r[v] * sc + rs;
#pragma unroll
            for (int t = 0; t < 4; ++t) acc[t][v] = acc[t][v] * sc;
            const int prow = v + (hi << 3);
            myp[prow * 32 + col]      = (__bf16)p0;
            myp[prow * 32 + 16 + col] = (__bf16)p1;
        }

        // ---- P (16x32) back as A-fragment; PV WMMAs --------------------
        v16bf pa = load_frag_a(myp, 32);
        v16bf bv[4];
#pragma unroll
        for (int t = 0; t < 4; ++t)
            bv[t] = load_frag_b(vc + (16 * t) * 32, 32);
#pragma unroll
        for (int t = 0; t < 4; ++t)
            acc[t] = wmma_bf16(pa, bv[t], acc[t]);

        __syncthreads();                 // all waves done reading cur buffer
    }

    // ---- epilogue: divide by l, store bf16 into (B, L, H, hd) ----------
    const int b = bh >> 4, h = bh & 15;
    const int row = m0 + (hi << 3);
#pragma unroll
    for (int v = 0; v < 8; ++v) {
        const float inv = 1.0f / l_r[v];
        const size_t obase = ((size_t)(b * L_SEQ + row + v) * N_HEAD + h) * HD;
#pragma unroll
        for (int t = 0; t < 4; ++t)
            O[obase + 16 * t + col] = (__bf16)(acc[t][v] * inv);
    }
}

// ---------------------------------------------------------------------------
// Host-side launcher
// ---------------------------------------------------------------------------
extern "C" void kernel_launch(void* const* d_in, const int* in_sizes, int n_in,
                              void* d_out, int out_size, void* d_ws, size_t ws_size,
                              hipStream_t stream) {
    (void)in_sizes; (void)n_in; (void)out_size; (void)ws_size;

    const float* x     = (const float*)d_in[0];   // (B,L,D)
    const float* W_qkv = (const float*)d_in[1];   // (3D, D)
    const float* w_qn  = (const float*)d_in[2];   // (hd,)
    const float* w_kn  = (const float*)d_in[3];   // (hd,)
    const float* W_out = (const float*)d_in[4];   // (D, D)
    float*       out   = (float*)d_out;           // (B,L,D) f32

    const int M  = B_SZ * L_SEQ;      // 4096
    const int K  = D_MOD;             // 1024
    const int N3 = 3 * D_MOD;         // 3072

    // workspace carve-out (~96 MB, fits comfortably in the 192 MB L2)
    char* p = (char*)d_ws;
    auto carve = [&](size_t bytes) { void* r = (void*)p; p += (bytes + 255) & ~(size_t)255; return r; };
    __bf16* xb    = (__bf16*)carve((size_t)M * K * 2);
    __bf16* wqkvb = (__bf16*)carve((size_t)N3 * K * 2);
    __bf16* woutb = (__bf16*)carve((size_t)K * K * 2);
    float*  qkvf  = (float* )carve((size_t)M * N3 * 4);
    __bf16* Qb    = (__bf16*)carve((size_t)M * D_MOD * 2);
    __bf16* Kb    = (__bf16*)carve((size_t)M * D_MOD * 2);
    __bf16* Vtb   = (__bf16*)carve((size_t)M * D_MOD * 2);
    __bf16* attnb = (__bf16*)carve((size_t)M * D_MOD * 2);

    // 1) f32 -> bf16 conversions
    {
        int n = M * K;
        cvt_f32_bf16<<<(n + 255) / 256, 256, 0, stream>>>(x, xb, n);
        n = N3 * K;
        cvt_f32_bf16<<<(n + 255) / 256, 256, 0, stream>>>(W_qkv, wqkvb, n);
        n = K * K;
        cvt_f32_bf16<<<(n + 255) / 256, 256, 0, stream>>>(W_out, woutb, n);
    }

    // 2) qkv = x @ W_qkv^T   (4096 x 3072 x 1024)
    gemm_bf16_nt<<<dim3(N3 / 256, M / 32), 256, 0, stream>>>(xb, wqkvb, qkvf, M, N3, K);

    // 3) RMSNorm q,k + scatter to Q/K/Vt bf16 layouts
    rmsnorm_scatter<<<M, 256, 0, stream>>>(qkvf, w_qn, w_kn, Qb, Kb, Vtb);

    // 4) flash attention per (b,h), async-LDS staged K/V
    flash_attn<<<dim3(L_SEQ / 128, B_SZ * N_HEAD), 256, 0, stream>>>(Qb, Kb, Vtb, attnb);

    // 5) out = attn @ W_out^T  (4096 x 1024 x 1024)
    gemm_bf16_nt<<<dim3(D_MOD / 256, M / 32), 256, 0, stream>>>(attnb, woutb, out, M, D_MOD, K);
}
